// TextGuidedCBAM_27599459844825
// MI455X (gfx1250) — matrix operand order repaired and checked by reading, not compiled, not run on hardware
//
#include <hip/hip_runtime.h>
#include <hip/hip_bf16.h>

// Problem constants
#define Bb   32
#define Cc   1024
#define Nn   784        // 28*28
#define TXT  768
#define Aa   256
#define REDd 64

typedef __attribute__((ext_vector_type(16))) _Float16 v16h;
typedef __attribute__((ext_vector_type(8)))  float    v8f;
typedef unsigned int u32;
typedef __attribute__((ext_vector_type(4))) u32 su4;   // D# group 0 (4 SGPRs)
typedef __attribute__((ext_vector_type(8))) u32 su8;   // D# group 1 (8 SGPRs)

#define WMMA_F16(a,b,c) __builtin_amdgcn_wmma_f32_16x16x32_f16(false,(a),false,(b),(short)0,(c),false,false)

// ---- WMMA operand loaders (contiguous k order: element e of lane half h holds k = 16h + e, wave32) ----
// A: 16x32 f16, row-major f32 source with leading dim lda.
__device__ __forceinline__ v16h load_a_rm(const float* A, int lda, int row0, int k0, int lane) {
    int m  = row0 + (lane & 15);
    int hi = (lane & 16) ? 16 : 0;
    const float* p = A + (size_t)m * lda + k0;
    v16h a;
#pragma unroll
    for (int e = 0; e < 8; ++e) {
        a[e]     = (_Float16)p[hi + e];
        a[8 + e] = (_Float16)p[hi + 8 + e];
    }
    return a;
}
// B[k][n] = W[n0+n][k]   (W row-major [N,K]; i.e. torch Linear weight, B = W^T)
__device__ __forceinline__ v16h load_b_nk(const float* W, int ldk, int n0, int k0, int lane) {
    int n  = n0 + (lane & 15);
    int kb = k0 + ((lane & 16) ? 16 : 0);
    const float* p = W + (size_t)n * ldk + kb;
    v16h b;
#pragma unroll
    for (int e = 0; e < 16; ++e) b[e] = (_Float16)p[e];
    return b;
}
// B[k][n] = W[k0+k][n0+n]  (W row-major [K,N])
__device__ __forceinline__ v16h load_b_kn(const float* W, int ldn, int n0, int k0, int lane) {
    int n  = n0 + (lane & 15);
    int kb = k0 + ((lane & 16) ? 16 : 0);
    v16h b;
#pragma unroll
    for (int e = 0; e < 16; ++e) b[e] = (_Float16)W[(size_t)(kb + e) * ldn + n];
    return b;
}
__device__ __forceinline__ void store_c(float* D, int ldn, int row0, int n0, int lane, v8f c) {
    int n  = n0 + (lane & 15);
    int mb = row0 + ((lane & 16) ? 8 : 0);
#pragma unroll
    for (int r = 0; r < 8; ++r) D[(size_t)(mb + r) * ldn + n] = c[r];
}

// ---------------- Pass 1: per-(b,c) avg + max over 784 pixels (1 wave/row) ----
__global__ void k_reduce(const float* __restrict__ x, float* __restrict__ avg, float* __restrict__ mx) {
    int row  = (blockIdx.x * blockDim.x + threadIdx.x) >> 5;   // b*C + c
    int lane = threadIdx.x & 31;
    const float* p = x + (size_t)row * Nn;
    float s = 0.f, m = -3.402823466e38f;
    for (int i = lane; i < Nn; i += 32) { float v = p[i]; s += v; m = fmaxf(m, v); }
#pragma unroll
    for (int off = 16; off > 0; off >>= 1) {
        s += __shfl_xor(s, off, 32);
        m  = fmaxf(m, __shfl_xor(m, off, 32));
    }
    if (lane == 0) { avg[row] = s * (1.0f / (float)Nn); mx[row] = m; }
}

// ---------------- MLP stage 1: H = relu([avg;mx] @ w1^T)  -> [64,64] ----------
__global__ void k_mlp1(const float* __restrict__ avg, const float* __restrict__ mx,
                       const float* __restrict__ w1, float* __restrict__ Hbuf) {
    int wid  = threadIdx.x >> 5;        // 0..15
    int lane = threadIdx.x & 31;
    int mt = wid >> 2, nt = wid & 3;
    const float* Z = (mt < 2) ? avg : mx;
    int row0 = (mt & 1) * 16;
    v8f c = {};
    for (int k = 0; k < Cc; k += 32) {
        v16h a = load_a_rm(Z, Cc, row0, k, lane);
        v16h b = load_b_nk(w1, Cc, nt * 16, k, lane);
        c = WMMA_F16(a, b, c);
    }
    int n  = nt * 16 + (lane & 15);
    int mb = mt * 16 + ((lane & 16) ? 8 : 0);
#pragma unroll
    for (int r = 0; r < 8; ++r) Hbuf[(mb + r) * REDd + n] = fmaxf(c[r], 0.0f);
}

// ---------------- MLP stage 2: G = H @ w2^T  -> [64,1024] ---------------------
__global__ void k_mlp2(const float* __restrict__ Hbuf, const float* __restrict__ w2,
                       float* __restrict__ G) {
    int wid  = blockIdx.x * (blockDim.x >> 5) + (threadIdx.x >> 5);   // 0..255
    int lane = threadIdx.x & 31;
    int mt = wid >> 6, nt = wid & 63;
    v8f c = {};
    for (int k = 0; k < REDd; k += 32) {
        v16h a = load_a_rm(Hbuf, REDd, mt * 16, k, lane);
        v16h b = load_b_nk(w2, REDd, nt * 16, k, lane);
        c = WMMA_F16(a, b, c);
    }
    store_c(G, Cc, mt * 16, nt * 16, lane, c);
}

// ---------------- ch_attn = sigmoid(G[b] + G[b+32]) ---------------------------
__global__ void k_chattn(const float* __restrict__ G, float* __restrict__ ch) {
    int i = blockIdx.x * 256 + threadIdx.x;                    // < 32768
    float v = G[i] + G[Bb * Cc + i];
    ch[i] = 1.0f / (1.0f + __expf(-v));
}

// ---------------- q = text_feat @ wq^T  -> [32,256] ---------------------------
__global__ void k_q(const float* __restrict__ text, const float* __restrict__ wq,
                    float* __restrict__ q) {
    int wid  = blockIdx.x * 8 + (threadIdx.x >> 5);            // 0..31
    int lane = threadIdx.x & 31;
    int mt = wid >> 4, nt = wid & 15;
    v8f c = {};
    for (int k = 0; k < TXT; k += 32) {
        v16h a = load_a_rm(text, TXT, mt * 16, k, lane);
        v16h b = load_b_nk(wq, TXT, nt * 16, k, lane);
        c = WMMA_F16(a, b, c);
    }
    store_c(q, Aa, mt * 16, nt * 16, lane, c);
}

// ---------------- qks = (q @ wk) * ch_attn / 16  -> [32,1024] -----------------
__global__ void k_qks(const float* __restrict__ q, const float* __restrict__ wk,
                      const float* __restrict__ ch, float* __restrict__ qks) {
    int wid  = blockIdx.x * 8 + (threadIdx.x >> 5);            // 0..127
    int lane = threadIdx.x & 31;
    int mt = wid >> 6, nt = wid & 63;
    v8f c = {};
    for (int k = 0; k < Aa; k += 32) {
        v16h a = load_a_rm(q, Aa, mt * 16, k, lane);
        v16h b = load_b_kn(wk, Cc, nt * 16, k, lane);          // wk row-major [A,C] = [K,N]
        c = WMMA_F16(a, b, c);
    }
    int n  = nt * 16 + (lane & 15);
    int mb = mt * 16 + ((lane & 16) ? 8 : 0);
#pragma unroll
    for (int r = 0; r < 8; ++r) {
        int idx = (mb + r) * Cc + n;
        qks[idx] = c[r] * ch[idx] * 0.0625f;                   // 1/sqrt(256)
    }
}

// ---------------- Pass 2: scores[b,n] = sum_c qks[b,c]*x[b,c,n] ---------------
// qks[b,:] (4 KB) is staged into LDS by the Tensor Data Mover:
//   tensor_load_to_lds D#(group0: 4 SGPRs, group1: 8 SGPRs), 1-D tile of 1024 f32.
__global__ void k_scores(const float* __restrict__ x, const float* __restrict__ qks,
                         float* __restrict__ scores) {
    __shared__ float w[Cc];
    int b = blockIdx.x >> 2;
    int n = (blockIdx.x & 3) * 256 + threadIdx.x;

    if (threadIdx.x < 32) {   // wave 0 issues the TDM transfer (EXEC-independent, per-wave TENSORcnt)
        unsigned long long ga = (unsigned long long)(const void*)(qks + (size_t)b * Cc);
        u32 ldsoff = (u32)(unsigned long long)(void*)&w[0];    // LDS byte offset = addr[31:0]
        su4 g0;
        g0.x = 1u;                                             // count=1, user descriptor
        g0.y = ldsoff;                                         // lds_addr
        g0.z = (u32)ga;                                        // global_addr[31:0]
        g0.w = (u32)((ga >> 32) & 0x1FFFFFFu) | (2u << 30);    // global_addr[56:32] | type=2
        su8 g1;
        g1[0] = 2u << 16;                                      // data_size = 4 bytes
        g1[1] = (u32)Cc << 16;                                 // tensor_dim0[15:0] in bits 63:48
        g1[2] = 1u << 16;                                      // tensor_dim0[31:16]=0, tensor_dim1=1
        g1[3] = (u32)Cc << 16;                                 // tile_dim0 = 1024
        g1[4] = 0u;                                            // tile_dim1/2 unused
        g1[5] = (u32)Cc;                                       // tensor_dim0_stride = 1024
        g1[6] = 0u;
        g1[7] = 0u;
        asm volatile("tensor_load_to_lds %0, %1" :: "s"(g0), "s"(g1) : "memory");
        __builtin_amdgcn_s_wait_tensorcnt(0);
    }
    __syncthreads();

    if (n < Nn) {
        const float* xp = x + (size_t)b * Cc * Nn + n;
        float acc = 0.f;
#pragma unroll 4
        for (int c = 0; c < Cc; ++c) acc += w[c] * xp[(size_t)c * Nn];
        scores[b * Nn + n] = acc;
    }
}

// ---------------- softmax over N per batch ------------------------------------
__global__ void k_softmax(const float* __restrict__ scores, float* __restrict__ attn) {
    __shared__ float red[256];
    int b = blockIdx.x, t = threadIdx.x;
    const float* s = scores + b * Nn;
    float m = -3.402823466e38f;
    for (int n = t; n < Nn; n += 256) m = fmaxf(m, s[n]);
    red[t] = m; __syncthreads();
    for (int off = 128; off > 0; off >>= 1) { if (t < off) red[t] = fmaxf(red[t], red[t + off]); __syncthreads(); }
    m = red[0]; __syncthreads();
    float sum = 0.f;
    for (int n = t; n < Nn; n += 256) { float e = __expf(s[n] - m); attn[b * Nn + n] = e; sum += e; }
    red[t] = sum; __syncthreads();
    for (int off = 128; off > 0; off >>= 1) { if (t < off) red[t] += red[t + off]; __syncthreads(); }
    float inv = 1.0f / red[0];
    for (int n = t; n < Nn; n += 256) attn[b * Nn + n] *= inv;
}

// ---------------- Pass 3: xattn[b,c] = ch[b,c]*sum_n attn[b,n]*x[b,c,n] -------
__global__ void k_xattn(const float* __restrict__ x, const float* __restrict__ attn,
                        const float* __restrict__ ch, float* __restrict__ xatt) {
    int row  = (blockIdx.x * blockDim.x + threadIdx.x) >> 5;   // b*C + c
    int lane = threadIdx.x & 31;
    int b = row >> 10;
    const float* xp = x + (size_t)row * Nn;
    const float* ap = attn + b * Nn;
    float s = 0.f;
    for (int i = lane; i < Nn; i += 32) s += ap[i] * xp[i];
#pragma unroll
    for (int off = 16; off > 0; off >>= 1) s += __shfl_xor(s, off, 32);
    if (lane == 0) xatt[row] = s * ch[row];
}

// ---------------- attended = xattn @ wv^T -> [32,256] -------------------------
__global__ void k_att(const float* __restrict__ xatt, const float* __restrict__ wv,
                      float* __restrict__ attended) {
    int wid  = blockIdx.x * 8 + (threadIdx.x >> 5);            // 0..31
    int lane = threadIdx.x & 31;
    int mt = wid >> 4, nt = wid & 15;
    v8f c = {};
    for (int k = 0; k < Cc; k += 32) {
        v16h a = load_a_rm(xatt, Cc, mt * 16, k, lane);
        v16h b = load_b_nk(wv, Cc, nt * 16, k, lane);
        c = WMMA_F16(a, b, c);
    }
    store_c(attended, Aa, mt * 16, nt * 16, lane, c);
}

// ---------------- spatial scalar = sigmoid(attended @ wo^T) -------------------
__global__ void k_sscalar(const float* __restrict__ attended, const float* __restrict__ wo,
                          float* __restrict__ ss) {
    int t = threadIdx.x;
    if (t < Bb) {
        float s = 0.f;
        for (int a = 0; a < Aa; ++a) s += attended[t * Aa + a] * wo[a];
        ss[t] = 1.0f / (1.0f + __expf(-s));
    }
}

// ---------------- Pass 4: out1 = x*ch*ss ; separate broadcast of ss -----------
__global__ void k_out(const float* __restrict__ x, const float* __restrict__ ch,
                      const float* __restrict__ ss, float* __restrict__ out) {
    int row = blockIdx.x;                                      // b*C + c
    int b = row >> 10;
    float sc = ch[row] * ss[b];
    const float* xp = x + (size_t)row * Nn;
    float* op = out + (size_t)row * Nn;
    for (int i = threadIdx.x; i < Nn; i += 256) op[i] = xp[i] * sc;
}
__global__ void k_out2(const float* __restrict__ ss, float* __restrict__ out2) {
    int idx = blockIdx.x * 256 + threadIdx.x;
    if (idx < Bb * Nn) out2[idx] = ss[idx / Nn];
}

extern "C" void kernel_launch(void* const* d_in, const int* in_sizes, int n_in,
                              void* d_out, int out_size, void* d_ws, size_t ws_size,
                              hipStream_t stream) {
    (void)in_sizes; (void)n_in; (void)out_size; (void)ws_size;
    const float* x    = (const float*)d_in[0];
    const float* text = (const float*)d_in[1];
    const float* w1   = (const float*)d_in[2];
    const float* w2   = (const float*)d_in[3];
    const float* wq   = (const float*)d_in[4];
    const float* wk   = (const float*)d_in[5];
    const float* wv   = (const float*)d_in[6];
    const float* wo   = (const float*)d_in[7];

    float* ws = (float*)d_ws;
    float* avg      = ws;                 // 32768
    float* mx       = avg + Bb * Cc;      // 32768
    float* ch       = mx + Bb * Cc;       // 32768
    float* Hbuf     = ch + Bb * Cc;       // 4096
    float* G        = Hbuf + 64 * REDd;   // 65536
    float* q        = G + 64 * Cc;        // 8192
    float* qks      = q + Bb * Aa;        // 32768
    float* scores   = qks + Bb * Cc;      // 25088
    float* attn     = scores + Bb * Nn;   // 25088
    float* xatt     = attn + Bb * Nn;     // 32768
    float* attended = xatt + Bb * Cc;     // 8192
    float* ss       = attended + Bb * Aa; // 32

    float* out1 = (float*)d_out;
    float* out2 = out1 + (size_t)Bb * Cc * Nn;

    k_reduce <<<4096, 256, 0, stream>>>(x, avg, mx);
    k_mlp1   <<<1, 512, 0, stream>>>(avg, mx, w1, Hbuf);
    k_mlp2   <<<32, 256, 0, stream>>>(Hbuf, w2, G);
    k_chattn <<<128, 256, 0, stream>>>(G, ch);
    k_q      <<<4, 256, 0, stream>>>(text, wq, q);
    k_qks    <<<16, 256, 0, stream>>>(q, wk, ch, qks);
    k_scores <<<128, 256, 0, stream>>>(x, qks, scores);
    k_softmax<<<32, 256, 0, stream>>>(scores, attn);
    k_xattn  <<<4096, 256, 0, stream>>>(x, attn, ch, xatt);
    k_att    <<<4, 256, 0, stream>>>(xatt, wv, attended);
    k_sscalar<<<1, 32, 0, stream>>>(attended, wo, ss);
    k_out    <<<Bb * Cc, 256, 0, stream>>>(x, ch, ss, out1);
    k_out2   <<<98, 256, 0, stream>>>(ss, out2);
}
